// SelfAttention_40441412059246
// MI455X (gfx1250) — matrix operand order repaired and checked by reading, not compiled, not run on hardware
//
#include <hip/hip_runtime.h>

#define EMBED 1024
#define HEADS 16
#define HDIM  64
#define NBATCH 4
#define SEQ   2048

typedef __attribute__((ext_vector_type(16))) __bf16 v16bf;
typedef __attribute__((ext_vector_type(8)))  __bf16 v8bf;
typedef __attribute__((ext_vector_type(8)))  float  v8f;

union AFrag { v16bf v; v8bf h[2]; __bf16 e[16]; };
union CAcc  { v8f   v; float  e[8]; };
union Stage { uint4 u[2]; __bf16 e[16]; };

__device__ __forceinline__ v8f wmma_bf16(v16bf a, v16bf b, v8f c) {
  return __builtin_amdgcn_wmma_f32_16x16x32_bf16(false, a, false, b, (short)0, c, false, false);
}

// Load a 16-element bf16 fragment as two contiguous 8-element (16B) runs.
__device__ __forceinline__ v16bf ld_frag(const __bf16* p0, const __bf16* p1) {
  AFrag f;
  f.h[0] = *(const v8bf*)p0;
  f.h[1] = *(const v8bf*)p1;
  return f.v;
}

// A-fragment from f32 source (two contiguous runs of 8 floats), convert to bf16.
__device__ __forceinline__ v16bf ld_afrag_f32(const float* p0, const float* p1) {
  AFrag f;
#pragma unroll
  for (int i = 0; i < 8; ++i) {
    f.e[i]     = (__bf16)p0[i];
    f.e[8 + i] = (__bf16)p1[i];
  }
  return f.v;
}

// ---------------------------------------------------------------- kernel 0
__global__ void cvt_f32_to_bf16(const float* __restrict__ src,
                                __bf16* __restrict__ dst, int n) {
  int i = blockIdx.x * blockDim.x + threadIdx.x;
  if (i < n) dst[i] = (__bf16)src[i];
}

// ---------------------------------------------------------------- kernel 1
// Per-head projections: {q,k,v}' [n][h][s][d] (bf16) = x[n][s][h*64+ :] @ W^T
__global__ __launch_bounds__(128) void proj_kernel(
    const float* __restrict__ xk, const float* __restrict__ xq,
    const float* __restrict__ xv,
    const __bf16* __restrict__ wk, const __bf16* __restrict__ wq,
    const __bf16* __restrict__ wv,
    __bf16* __restrict__ kp, __bf16* __restrict__ qp, __bf16* __restrict__ vp) {
  const int lane = threadIdx.x & 31;
  const int wave = threadIdx.x >> 5;
  const int tile = blockIdx.x * 4 + wave;     // 8192 tiles = N*H*(S/16)
  const int n  = tile >> 11;                  // / (HEADS * SEQ/16)
  const int h  = (tile >> 7) & 15;
  const int s0 = (tile & 127) << 4;
  const int m = lane & 15, g = lane >> 4;

  const float*  xin[3] = {xq, xk, xv};
  const __bf16* win[3] = {wq, wk, wv};
  __bf16*       out[3] = {qp, kp, vp};

#pragma unroll
  for (int t = 0; t < 3; ++t) {
    const float* xb = xin[t] + ((size_t)n * SEQ + s0) * EMBED + h * HDIM;
    CAcc acc[4];
#pragma unroll
    for (int d = 0; d < 4; ++d)
#pragma unroll
      for (int i = 0; i < 8; ++i) acc[d].e[i] = 0.f;

#pragma unroll
    for (int kc = 0; kc < 2; ++kc) {
      const float* ap = xb + (size_t)m * EMBED + kc * 32 + g * 8;
      v16bf a = ld_afrag_f32(ap, ap + 16);
#pragma unroll
      for (int d = 0; d < 4; ++d) {
        // B[k][nn] = W[d*16+nn][k] : row nn of W, k contiguous
        const __bf16* bp = win[t] + (size_t)(d * 16 + m) * HDIM + kc * 32 + g * 8;
        v16bf b = ld_frag(bp, bp + 16);
        acc[d].v = wmma_bf16(a, b, acc[d].v);
      }
    }
    __bf16* ob = out[t] + (((size_t)n * HEADS + h) * SEQ + s0) * HDIM;
#pragma unroll
    for (int d = 0; d < 4; ++d)
#pragma unroll
      for (int r = 0; r < 8; ++r)
        ob[(size_t)(r + 8 * g) * HDIM + d * 16 + m] = (__bf16)acc[d].e[r];
  }
}

// ---------------------------------------------------------------- kernel 2
// Flash attention. Block = 4 waves x 32 q-rows = 128 q-rows of one (n,h).
// Double-buffered K/V staging through LDS; global prefetch 2 chunks ahead.
// scale = 1/(EMBED/2) = 1/512 (faithful to reference quirk).
__global__ __launch_bounds__(128) void attn_kernel(
    const __bf16* __restrict__ qp, const __bf16* __restrict__ kp,
    const __bf16* __restrict__ vp, __bf16* __restrict__ att) {
  __shared__ __attribute__((aligned(16))) __bf16 ldsK[2 * 32 * 72];      // [buf][key][d] stride 72
  __shared__ __attribute__((aligned(16))) __bf16 ldsV[2 * 64 * 40];      // [buf][d][key] stride 40
  __shared__ __attribute__((aligned(16))) __bf16 ldsP[4 * 2 * 16 * 40];  // [wave][tile] P transpose

  const int tid  = threadIdx.x;
  const int lane = tid & 31, wave = tid >> 5;
  const int b  = blockIdx.x;                 // 1024 = N*H*(S/128)
  const int n  = b >> 8;
  const int h  = (b >> 4) & 15;
  const int sq = (b & 15) * 128 + wave * 32; // two 16-row tiles: sq, sq+16
  const int m = lane & 15, g = lane >> 4;

  const size_t nh = ((size_t)n * HEADS + h) * SEQ;
  const __bf16* qb = qp + nh * HDIM;
  const __bf16* kb = kp + nh * HDIM;
  const __bf16* vb = vp + nh * HDIM;

  // Q fragments resident in VGPRs for the whole K/V sweep
  v16bf qa[2][2];
#pragma unroll
  for (int t = 0; t < 2; ++t)
#pragma unroll
    for (int kc = 0; kc < 2; ++kc) {
      const __bf16* p = qb + (size_t)(sq + t * 16 + m) * HDIM + kc * 32 + g * 8;
      qa[t][kc] = ld_frag(p, p + 16);
    }

  CAcc o[2][4];
  float mrow[2][8], lrow[2][8];
#pragma unroll
  for (int t = 0; t < 2; ++t) {
#pragma unroll
    for (int d = 0; d < 4; ++d)
#pragma unroll
      for (int i = 0; i < 8; ++i) o[t][d].e[i] = 0.f;
#pragma unroll
    for (int r = 0; r < 8; ++r) { mrow[t][r] = -1e30f; lrow[t][r] = 0.f; }
  }

  // staging addresses for this thread
  const int keyK = tid >> 2, segK = tid & 3;   // K: 32 keys x 4 segs of 16 bf16
  const int keyV = tid & 31, dsegV = tid >> 5; // V: 32 keys x 4 d-segs of 16
  const __bf16* ksrc = kb + (size_t)keyK * HDIM + segK * 16;
  const __bf16* vsrc = vb + (size_t)keyV * HDIM + dsegV * 16;
  const int CH = 32 * HDIM;                    // global elems per chunk

  // preload chunk 0 and stage into buffer 0
  Stage sk, sv;
  sk.u[0] = *(const uint4*)ksrc;          sk.u[1] = *(const uint4*)(ksrc + 8);
  sv.u[0] = *(const uint4*)vsrc;          sv.u[1] = *(const uint4*)(vsrc + 8);
  {
    __bf16* kd = &ldsK[keyK * 72 + segK * 16];
    *(uint4*)kd = sk.u[0]; *(uint4*)(kd + 8) = sk.u[1];
#pragma unroll
    for (int i = 0; i < 16; ++i) ldsV[(dsegV * 16 + i) * 40 + keyV] = sv.e[i];
  }
  __syncthreads();

  const int NCH = SEQ / 32;
  for (int c = 0; c < NCH; ++c) {
    const int buf = c & 1;
    const __bf16* Kb = &ldsK[buf * 32 * 72];
    const __bf16* Vb = &ldsV[buf * 64 * 40];

    // issue next chunk's global loads now (latency hidden under WMMAs)
    if (c + 1 < NCH) {
      sk.u[0] = *(const uint4*)(ksrc + (size_t)(c + 1) * CH);
      sk.u[1] = *(const uint4*)(ksrc + (size_t)(c + 1) * CH + 8);
      sv.u[0] = *(const uint4*)(vsrc + (size_t)(c + 1) * CH);
      sv.u[1] = *(const uint4*)(vsrc + (size_t)(c + 1) * CH + 8);
    }
    if (c + 2 < NCH) {          // gfx1250 global_prefetch_b8 two chunks ahead
      __builtin_prefetch(ksrc + (size_t)(c + 2) * CH, 0, 3);
      __builtin_prefetch(vsrc + (size_t)(c + 2) * CH, 0, 3);
    }

    // --- energy: E[t] = Q_t(16x64) . K^T ; B fragments shared by both tiles ---
    CAcc e[2][2];
#pragma unroll
    for (int t = 0; t < 2; ++t)
#pragma unroll
      for (int i = 0; i < 8; ++i) { e[t][0].e[i] = 0.f; e[t][1].e[i] = 0.f; }
#pragma unroll
    for (int kc = 0; kc < 2; ++kc) {
      const __bf16* b0 = &Kb[m * 72 + kc * 32 + g * 8];
      v16bf bk0 = ld_frag(b0, b0 + 16);
      const __bf16* b1 = &Kb[(m + 16) * 72 + kc * 32 + g * 8];
      v16bf bk1 = ld_frag(b1, b1 + 16);
#pragma unroll
      for (int t = 0; t < 2; ++t) {
        e[t][0].v = wmma_bf16(qa[t][kc], bk0, e[t][0].v);
        e[t][1].v = wmma_bf16(qa[t][kc], bk1, e[t][1].v);
      }
    }

    // --- online softmax + P transpose, per q-tile ---
    v16bf pa[2];
#pragma unroll
    for (int t = 0; t < 2; ++t) {
      const float sc = 1.0f / 512.0f;
      float mx[8];
#pragma unroll
      for (int r = 0; r < 8; ++r) {
        e[t][0].e[r] *= sc; e[t][1].e[r] *= sc;
        mx[r] = fmaxf(e[t][0].e[r], e[t][1].e[r]);
      }
#pragma unroll
      for (int off = 1; off < 16; off <<= 1)
#pragma unroll
        for (int r = 0; r < 8; ++r)
          mx[r] = fmaxf(mx[r], __shfl_xor(mx[r], off, 32));

      float p0[8], p1[8], rs[8], alpha[8];
#pragma unroll
      for (int r = 0; r < 8; ++r) {
        float mn = fmaxf(mrow[t][r], mx[r]);
        alpha[r]   = __expf(mrow[t][r] - mn);
        mrow[t][r] = mn;
        p0[r] = __expf(e[t][0].e[r] - mn);
        p1[r] = __expf(e[t][1].e[r] - mn);
        rs[r] = p0[r] + p1[r];
      }
#pragma unroll
      for (int off = 1; off < 16; off <<= 1)
#pragma unroll
        for (int r = 0; r < 8; ++r) rs[r] += __shfl_xor(rs[r], off, 32);
#pragma unroll
      for (int r = 0; r < 8; ++r) {
        lrow[t][r] = lrow[t][r] * alpha[r] + rs[r];
#pragma unroll
        for (int d = 0; d < 4; ++d) o[t][d].e[r] *= alpha[r];
      }

      // transpose P(16x32) through per-wave LDS into an A-fragment
      __bf16* pl = &ldsP[(wave * 2 + t) * 16 * 40];
#pragma unroll
      for (int r = 0; r < 8; ++r) {
        pl[(r + 8 * g) * 40 + m]      = (__bf16)p0[r];
        pl[(r + 8 * g) * 40 + 16 + m] = (__bf16)p1[r];
      }
      // LDS ops are in-order within a wave; read back immediately.
      const __bf16* pr = pl + m * 40 + g * 8;
      pa[t] = ld_frag(pr, pr + 16);
    }

    // --- O[t] += P[t] . V ; V fragments shared by both tiles ---
#pragma unroll
    for (int d = 0; d < 4; ++d) {
      const __bf16* vbp = &Vb[(d * 16 + m) * 40 + g * 8];
      v16bf bv = ld_frag(vbp, vbp + 16);
      o[0][d].v = wmma_bf16(pa[0], bv, o[0][d].v);
      o[1][d].v = wmma_bf16(pa[1], bv, o[1][d].v);
    }

    // --- stage next chunk into alternate buffer ---
    if (c + 1 < NCH) {
      __bf16* kd = &ldsK[(buf ^ 1) * 32 * 72 + keyK * 72 + segK * 16];
      *(uint4*)kd = sk.u[0]; *(uint4*)(kd + 8) = sk.u[1];
      __bf16* vd = &ldsV[(buf ^ 1) * 64 * 40];
#pragma unroll
      for (int i = 0; i < 16; ++i) vd[(dsegV * 16 + i) * 40 + keyV] = sv.e[i];
    }
    __syncthreads();
  }

  // --- normalize and store concat-head layout [n][s][h*64+d] as bf16 ---
#pragma unroll
  for (int t = 0; t < 2; ++t) {
    __bf16* ob = att + ((size_t)n * SEQ + sq + t * 16) * EMBED + h * HDIM;
#pragma unroll
    for (int d = 0; d < 4; ++d)
#pragma unroll
      for (int r = 0; r < 8; ++r)
        ob[(size_t)(r + 8 * g) * EMBED + d * 16 + m] =
            (__bf16)(o[t][d].e[r] / lrow[t][r]);
  }
}

// ---------------------------------------------------------------- kernel 3
// out[m][e] = att[m][:] @ Wo^T + bo  (M=8192, N=K=1024), f32 output
__global__ __launch_bounds__(128) void outproj_kernel(
    const __bf16* __restrict__ att, const __bf16* __restrict__ wo,
    const float* __restrict__ bo, float* __restrict__ out) {
  const int lane = threadIdx.x & 31, wave = threadIdx.x >> 5;
  const int tile = blockIdx.x * 4 + wave;   // 8192 = 512 row-tiles x 16 col-tiles
  const int mt = tile >> 4;
  const int nt = tile & 15;
  const int m = lane & 15, g = lane >> 4;

  const __bf16* ab = att + (size_t)(mt * 16 + m) * EMBED;
  CAcc acc[4];
#pragma unroll
  for (int d = 0; d < 4; ++d)
#pragma unroll
    for (int i = 0; i < 8; ++i) acc[d].e[i] = 0.f;

  for (int kc = 0; kc < EMBED / 32; ++kc) {
    const __bf16* ap = ab + kc * 32 + g * 8;
    v16bf a = ld_frag(ap, ap + 16);
#pragma unroll
    for (int d = 0; d < 4; ++d) {
      const __bf16* bp = wo + (size_t)(nt * 64 + d * 16 + m) * EMBED + kc * 32 + g * 8;
      acc[d].v = wmma_bf16(a, ld_frag(bp, bp + 16), acc[d].v);
    }
  }

  float* ob = out + (size_t)(mt * 16) * EMBED + nt * 64;
#pragma unroll
  for (int d = 0; d < 4; ++d) {
    float bias = bo[nt * 64 + d * 16 + m];
#pragma unroll
    for (int r = 0; r < 8; ++r)
      ob[(size_t)(r + 8 * g) * EMBED + d * 16 + m] = acc[d].e[r] + bias;
  }
}

// ---------------------------------------------------------------- launch
extern "C" void kernel_launch(void* const* d_in, const int* in_sizes, int n_in,
                              void* d_out, int out_size, void* d_ws, size_t ws_size,
                              hipStream_t stream) {
  (void)in_sizes; (void)n_in; (void)out_size; (void)ws_size;
  const float* keys  = (const float*)d_in[0];
  const float* query = (const float*)d_in[1];
  const float* vals  = (const float*)d_in[2];
  const float* Wk    = (const float*)d_in[3];
  const float* Wq    = (const float*)d_in[4];
  const float* Wv    = (const float*)d_in[5];
  const float* Wo    = (const float*)d_in[6];
  const float* bo    = (const float*)d_in[7];
  float* out = (float*)d_out;

  // workspace layout (bf16 elements)
  __bf16* ws = (__bf16*)d_ws;
  const size_t W64 = 64 * 64;           // 4096
  const size_t WOE = (size_t)EMBED * EMBED;
  const size_t PHD = (size_t)NBATCH * HEADS * SEQ * HDIM;   // 8,388,608
  __bf16* wq_bf = ws;
  __bf16* wk_bf = wq_bf + W64;
  __bf16* wv_bf = wk_bf + W64;
  __bf16* wo_bf = wv_bf + W64;
  __bf16* qp    = wo_bf + WOE;
  __bf16* kp    = qp + PHD;
  __bf16* vp    = kp + PHD;
  __bf16* att   = vp + PHD;             // [n][s][EMBED]

  cvt_f32_to_bf16<<<(int)((W64 + 255) / 256), 256, 0, stream>>>(Wq, wq_bf, (int)W64);
  cvt_f32_to_bf16<<<(int)((W64 + 255) / 256), 256, 0, stream>>>(Wk, wk_bf, (int)W64);
  cvt_f32_to_bf16<<<(int)((W64 + 255) / 256), 256, 0, stream>>>(Wv, wv_bf, (int)W64);
  cvt_f32_to_bf16<<<(int)((WOE + 255) / 256), 256, 0, stream>>>(Wo, wo_bf, (int)WOE);

  proj_kernel<<<2048, 128, 0, stream>>>(keys, query, vals, wk_bf, wq_bf, wv_bf,
                                        kp, qp, vp);
  attn_kernel<<<1024, 128, 0, stream>>>(qp, kp, vp, att);
  outproj_kernel<<<2048, 128, 0, stream>>>(att, wo_bf, bo, out);
}
